// SF_RCP_55336358641942
// MI455X (gfx1250) — matrix operand order repaired and checked by reading, not compiled, not run on hardware
//
#include <hip/hip_runtime.h>
#include <hip/hip_bf16.h>

typedef __attribute__((ext_vector_type(16))) _Float16 v16h;
typedef __attribute__((ext_vector_type(8)))  _Float16 v8h;
typedef __attribute__((ext_vector_type(8)))  float    v8f;

#define KNN_MAXN 8192

// ---------------- channel-first [B,C,N] -> row-major [B,N,C] ----------------
__global__ void k_chw_to_hwc(const float* __restrict__ in, float* __restrict__ out,
                             int B, int C, int N) {
    size_t t = (size_t)blockIdx.x * blockDim.x + threadIdx.x;
    if (t >= (size_t)B * N) return;
    int b = (int)(t / N), n = (int)(t % N);
    for (int c = 0; c < C; ++c)
        out[((size_t)b * N + n) * C + c] = in[((size_t)b * C + c) * N + n];
}

// ---------------- farthest point sampling: one block per batch, all-LDS ----------------
// dynamic LDS: sx[N] sy[N] sz[N] sdist[N]  (N=8192 -> 128 KB, fits 320 KB WGP LDS)
__global__ void k_fps(const float* __restrict__ xyz, int* __restrict__ fidx,
                      int N, int S) {
    extern __shared__ float smem[];
    float* sx = smem;
    float* sy = sx + N;
    float* sz = sy + N;
    float* sdist = sz + N;
    __shared__ float wbest[32];
    __shared__ int   wbesti[32];
    __shared__ int   sfar;
    int b = blockIdx.x;
    const float* px = xyz + (size_t)b * N * 3;
    int t = threadIdx.x, nt = blockDim.x;
    for (int i = t; i < N; i += nt) {
        sx[i] = px[i*3+0]; sy[i] = px[i*3+1]; sz[i] = px[i*3+2];
        sdist[i] = 1e10f;
    }
    if (t == 0) sfar = 0;
    __syncthreads();
    for (int it = 0; it < S; ++it) {
        int far = sfar;
        if (t == 0) fidx[(size_t)b * S + it] = far;
        float cx = sx[far], cy = sy[far], cz = sz[far];
        float best = -1.0f; int besti = 0;
        for (int i = t; i < N; i += nt) {
            float dx = sx[i]-cx, dy = sy[i]-cy, dz = sz[i]-cz;
            float d = fminf(sdist[i], dx*dx + dy*dy + dz*dz);
            sdist[i] = d;
            if (d > best) { best = d; besti = i; }   // ascending i: low index wins ties
        }
        for (int off = 16; off > 0; off >>= 1) {     // wave32 argmax
            float ob = __shfl_down(best, off, 32);
            int   oi = __shfl_down(besti, off, 32);
            if (ob > best || (ob == best && oi < besti)) { best = ob; besti = oi; }
        }
        int wid = t >> 5, lane = t & 31;
        if (lane == 0) { wbest[wid] = best; wbesti[wid] = besti; }
        __syncthreads();
        if (t < 32) {                                 // wave 0 reduces the 32 partials
            float bb = wbest[t]; int bi = wbesti[t];
            for (int off = 16; off > 0; off >>= 1) {
                float ob = __shfl_down(bb, off, 32);
                int   oi = __shfl_down(bi, off, 32);
                if (ob > bb || (ob == bb && oi < bi)) { bb = ob; bi = oi; }
            }
            if (t == 0) sfar = bi;
        }
        __syncthreads();
    }
}

// ---------------- gather sampled centers; also emit channel-first output ----------------
__global__ void k_gather_xyz(const float* __restrict__ xyz, const int* __restrict__ fidx,
                             float* __restrict__ nxyz, float* __restrict__ outCF,
                             int B, int N, int S) {
    size_t t = (size_t)blockIdx.x * blockDim.x + threadIdx.x;
    if (t >= (size_t)B * S) return;
    int b = (int)(t / S), s = (int)(t % S);
    int id = fidx[t];
    const float* p = xyz + ((size_t)b * N + id) * 3;
    float* o = nxyz + t * 3;
    o[0] = p[0]; o[1] = p[1]; o[2] = p[2];
    if (outCF) {
        outCF[((size_t)b*3 + 0) * S + s] = p[0];
        outCF[((size_t)b*3 + 1) * S + s] = p[1];
        outCF[((size_t)b*3 + 2) * S + s] = p[2];
    }
}

// ---------------- kNN: one wave per query, LDS-cached distances ----------------
__global__ void k_knn(const float* __restrict__ query, const float* __restrict__ pts,
                      int* __restrict__ nidx, int S, int N, int K) {
    __shared__ float sd[KNN_MAXN];
    int q = blockIdx.x;          // b*S + s
    int b = q / S;
    int lane = threadIdx.x;
    const float* P = pts + (size_t)b * N * 3;
    float qx = query[(size_t)q*3+0], qy = query[(size_t)q*3+1], qz = query[(size_t)q*3+2];
    for (int i = lane; i < N; i += 32) {
        float dx = P[i*3+0]-qx, dy = P[i*3+1]-qy, dz = P[i*3+2]-qz;
        sd[i] = dx*dx + dy*dy + dz*dz;
    }
    __syncthreads();
    float lastd = -1e30f; int lasti = -1;
    for (int j = 0; j < K; ++j) {
        float best = 1e30f; int besti = N;
        for (int i = lane; i < N; i += 32) {
            float d = sd[i];
            bool gt = (d > lastd) || (d == lastd && i > lasti);
            if (gt && (d < best || (d == best && i < besti))) { best = d; besti = i; }
        }
        for (int off = 16; off > 0; off >>= 1) {   // butterfly: all lanes converge
            float ob = __shfl_xor(best, off, 32);
            int   oi = __shfl_xor(besti, off, 32);
            if (ob < best || (ob == best && oi < besti)) { best = ob; besti = oi; }
        }
        if (lane == 0) nidx[(size_t)q * K + j] = besti;
        lastd = best; lasti = besti;
    }
}

// ---------------- build grouped input as padded f16 [row, Cpad] ----------------
__global__ void k_group_f16(const float* __restrict__ xyz, const float* __restrict__ feat,
                            const float* __restrict__ nxyz, const int* __restrict__ nidx,
                            _Float16* __restrict__ g,
                            int B, int S, int N, int K, int Cf, int Cpad) {
    size_t r = (size_t)blockIdx.x * blockDim.x + threadIdx.x;
    size_t rows = (size_t)B * S * K;
    if (r >= rows) return;
    int b = (int)(r / ((size_t)S * K));
    int s = (int)((r % ((size_t)S * K)) / K);
    int idx = nidx[r];
    const float* P = xyz  + ((size_t)b * N + idx) * 3;
    const float* Q = nxyz + ((size_t)b * S + s) * 3;
    _Float16* out = g + r * (size_t)Cpad;
    out[0] = (_Float16)(P[0]-Q[0]);
    out[1] = (_Float16)(P[1]-Q[1]);
    out[2] = (_Float16)(P[2]-Q[2]);
    const float* F = feat + ((size_t)b * N + idx) * Cf;
    for (int c = 0; c < Cf; ++c) out[3+c] = (_Float16)F[c];
    for (int c = 3 + Cf; c < Cpad; ++c) out[c] = (_Float16)0.0f;
}

// ---------------- weights f32 [Cout,Cin] -> f16 [Cout,Kpad] zero-padded ----------------
__global__ void k_w_to_f16(const float* __restrict__ W, _Float16* __restrict__ Wh,
                           int Cout, int Cin, int Kpad) {
    int t = blockIdx.x * blockDim.x + threadIdx.x;
    if (t >= Cout * Kpad) return;
    int n = t / Kpad, k = t % Kpad;
    Wh[t] = (_Float16)((k < Cin) ? W[n * Cin + k] : 0.0f);
}

// ---------------- IN + ReLU + convert to f16 activations ----------------
__global__ void k_act_f16(const float* __restrict__ Y, const float* __restrict__ mean,
                          const float* __restrict__ rstd, _Float16* __restrict__ Ah,
                          long long total, int C, int rowsPerBatch) {
    size_t t = (size_t)blockIdx.x * blockDim.x + threadIdx.x;
    if (t >= (size_t)total) return;
    int c = (int)(t % C);
    int b = (int)((t / C) / rowsPerBatch);
    float v = (Y[t] - mean[b * C + c]) * rstd[b * C + c];
    Ah[t] = (_Float16)(v > 0.0f ? v : 0.0f);
}

// ---------------- GEMM: Y[rows,COUT] = Ah[rows,KPAD] @ Wh[COUT,KPAD]^T + bias ----------
// Each wave computes TM 16-row tiles x full COUT; B fragments hoisted over the M loop.
// Fragment loads are contiguous b128s matching ISA 7.12.2 wave32 layouts.
template<int COUT, int KPAD, int TM>
__global__ void k_gemm_wmma(const _Float16* __restrict__ Ah,
                            const _Float16* __restrict__ Wh,
                            const float* __restrict__ bias,
                            float* __restrict__ Y, int rows) {
    constexpr int TN = COUT / 16;
    int lane = threadIdx.x & 31;
    int wid  = (int)(((size_t)blockIdx.x * blockDim.x + threadIdx.x) >> 5);
    int tiles = rows / (16 * TM);
    if (wid >= tiles) return;                // wave-uniform guard, EXEC all-ones inside
    int lo = (lane >= 16) ? 8  : 0;          // A: lane-half K offset
    int kb = (lane >= 16) ? 8  : 0;          // B: lane-half K offset
    const _Float16* abase = Ah + ((size_t)(wid * (16 * TM) + (lane & 15))) * KPAD + lo;
    v8f acc[TM][TN];
#pragma unroll
    for (int m = 0; m < TM; ++m)
#pragma unroll
        for (int t = 0; t < TN; ++t) acc[m][t] = (v8f){};
#pragma unroll
    for (int k0 = 0; k0 < KPAD; k0 += 32) {
        v16h bf[TN];
#pragma unroll
        for (int t = 0; t < TN; ++t) {       // B fragments: once per k-step per wave
            const _Float16* wrow = Wh + ((size_t)(t * 16 + (lane & 15))) * KPAD + k0 + kb;
            v8h blo = *(const v8h*)(wrow);
            v8h bhi = *(const v8h*)(wrow + 16);
            bf[t] = __builtin_shufflevector(blo, bhi,
                     0,1,2,3,4,5,6,7,8,9,10,11,12,13,14,15);
        }
#pragma unroll
        for (int m = 0; m < TM; ++m) {
            const _Float16* arow = abase + (size_t)m * 16 * KPAD + k0;
            v8h alo = *(const v8h*)(arow);        // K = k0+lo    .. +7
            v8h ahi = *(const v8h*)(arow + 16);   // K = k0+16+lo .. +7
            v16h a = __builtin_shufflevector(alo, ahi,
                      0,1,2,3,4,5,6,7,8,9,10,11,12,13,14,15);
#pragma unroll
            for (int t = 0; t < TN; ++t)
                acc[m][t] = __builtin_amdgcn_wmma_f32_16x16x32_f16(false, a, false, bf[t],
                                                                   (short)0, acc[m][t],
                                                                   false, false);
        }
    }
#pragma unroll
    for (int m = 0; m < TM; ++m) {
        int mBase = wid * (16 * TM) + m * 16 + ((lane >= 16) ? 8 : 0);
#pragma unroll
        for (int t = 0; t < TN; ++t) {
            int n = t * 16 + (lane & 15);
            float bs = bias[n];
#pragma unroll
            for (int j = 0; j < 8; ++j)
                Y[(size_t)(mBase + j) * COUT + n] = acc[m][t][j] + bs;
        }
    }
}

// ---------------- per-(batch,channel) instance-norm statistics ----------------
__global__ void k_stats(const float* __restrict__ Y, float* __restrict__ mean,
                        float* __restrict__ rstd, int Sk, int C) {
    int bc = blockIdx.x;                    // b*C + c
    int b = bc / C, ch = bc % C;
    const float* p = Y + (size_t)b * Sk * C + ch;
    __shared__ float ss[256], sq[256];
    float s = 0.0f, q = 0.0f;
    for (int i = threadIdx.x; i < Sk; i += blockDim.x) {
        float v = p[(size_t)i * C];
        s += v; q += v * v;
    }
    ss[threadIdx.x] = s; sq[threadIdx.x] = q;
    __syncthreads();
    for (int off = blockDim.x >> 1; off > 0; off >>= 1) {
        if ((int)threadIdx.x < off) {
            ss[threadIdx.x] += ss[threadIdx.x + off];
            sq[threadIdx.x] += sq[threadIdx.x + off];
        }
        __syncthreads();
    }
    if (threadIdx.x == 0) {
        float m = ss[0] / (float)Sk;
        float v = sq[0] / (float)Sk - m * m;
        mean[bc] = m;
        rstd[bc] = rsqrtf(v + 1e-5f);
    }
}

// ---------------- final IN + ReLU + max-pool over k ----------------
__global__ void k_nrm_pool(const float* __restrict__ Y, const float* __restrict__ mean,
                           const float* __restrict__ rstd,
                           float* __restrict__ outRM,   // [B,S,C] or null
                           float* __restrict__ outCF,   // [B,C,S] or null
                           int B, int S, int K, int C) {
    size_t t = (size_t)blockIdx.x * blockDim.x + threadIdx.x;
    if (t >= (size_t)B * S * C) return;
    int ch = (int)(t % C);
    int s  = (int)((t / C) % S);
    int b  = (int)(t / ((size_t)S * C));
    const float* p = Y + ((size_t)b * S + s) * K * C + ch;
    float m = mean[b * C + ch], r = rstd[b * C + ch];
    float mx = 0.0f;                        // ReLU output >= 0
    for (int j = 0; j < K; ++j) {
        float v = (p[(size_t)j * C] - m) * r;
        v = v > 0.0f ? v : 0.0f;
        mx = fmaxf(mx, v);
    }
    if (outRM) outRM[t] = mx;
    if (outCF) outCF[((size_t)b * C + ch) * S + s] = mx;
}

static inline int ceilDiv(long long a, long long b) { return (int)((a + b - 1) / b); }

extern "C" void kernel_launch(void* const* d_in, const int* in_sizes, int n_in,
                              void* d_out, int out_size, void* d_ws, size_t ws_size,
                              hipStream_t stream) {
    const int B = 4, N1 = 8192, S1 = 4096, S2 = 2048, K = 32;
    const long long rows1 = (long long)B * S1 * K;   // 524288
    const long long rows2 = (long long)B * S2 * K;   // 262144

    const float* pc      = (const float*)d_in[0];
    const float* feature = (const float*)d_in[1];
    const float* w1a = (const float*)d_in[2];  const float* b1a = (const float*)d_in[3];
    const float* w1b = (const float*)d_in[4];  const float* b1b = (const float*)d_in[5];
    const float* w1c = (const float*)d_in[6];  const float* b1c = (const float*)d_in[7];
    const float* w2a = (const float*)d_in[8];  const float* b2a = (const float*)d_in[9];
    const float* w2b = (const float*)d_in[10]; const float* b2b = (const float*)d_in[11];
    const float* w2c = (const float*)d_in[12]; const float* b2c = (const float*)d_in[13];

    // output layout: pc1 [B,3,S1], pc2 [B,3,S2], f2 [B,64,S2], idx1 [B,S1], idx2 [B,S2]
    float* out_pc1 = (float*)d_out;
    float* out_pc2 = out_pc1 + (size_t)B * 3 * S1;
    float* out_f2  = out_pc2 + (size_t)B * 3 * S2;
    int*   out_idx1 = (int*)(out_f2 + (size_t)B * 64 * S2);
    int*   out_idx2 = out_idx1 + (size_t)B * S1;

    // workspace allocator
    char* wp = (char*)d_ws;
    auto alloc = [&](size_t bytes) -> void* {
        void* p = (void*)wp;
        wp += (bytes + 255) & ~(size_t)255;
        return p;
    };
    float*    xyz   = (float*)alloc((size_t)B * N1 * 3 * 4);
    float*    fea   = (float*)alloc((size_t)B * N1 * 3 * 4);
    float*    nxyz1 = (float*)alloc((size_t)B * S1 * 3 * 4);
    int*      nidx1 = (int*)  alloc((size_t)B * S1 * K * 4);
    float*    f1    = (float*)alloc((size_t)B * S1 * 32 * 4);
    float*    nxyz2 = (float*)alloc((size_t)B * S2 * 3 * 4);
    int*      nidx2 = (int*)  alloc((size_t)B * S2 * K * 4);
    float*    meanb = (float*)alloc((size_t)B * 64 * 4);
    float*    rstdb = (float*)alloc((size_t)B * 64 * 4);
    _Float16* Wh    = (_Float16*)alloc((size_t)64 * 64 * 2);
    float*    bufY  = (float*)alloc((size_t)rows1 * 32 * 4);     // 64 MB (== rows2*64*4)
    _Float16* bufH  = (_Float16*)alloc((size_t)rows1 * 32 * 2);  // 32 MB (== rows2*64*2)
    (void)in_sizes; (void)n_in; (void)out_size; (void)ws_size;

    // ---------------- SA1 ----------------
    k_chw_to_hwc<<<ceilDiv((long long)B*N1,256),256,0,stream>>>(pc, xyz, B, 3, N1);
    k_chw_to_hwc<<<ceilDiv((long long)B*N1,256),256,0,stream>>>(feature, fea, B, 3, N1);
    k_fps<<<B,1024,(size_t)N1*16,stream>>>(xyz, out_idx1, N1, S1);     // 128 KB LDS
    k_gather_xyz<<<ceilDiv((long long)B*S1,256),256,0,stream>>>(xyz, out_idx1, nxyz1, out_pc1, B, N1, S1);
    k_knn<<<B*S1,32,0,stream>>>(nxyz1, xyz, nidx1, S1, N1, K);
    k_group_f16<<<ceilDiv(rows1,256),256,0,stream>>>(xyz, fea, nxyz1, nidx1, bufH, B, S1, N1, K, 3, 32);

    k_w_to_f16<<<ceilDiv(32*32,256),256,0,stream>>>(w1a, Wh, 32, 6, 32);
    k_gemm_wmma<32,32,4><<<ceilDiv(rows1/64,8),256,0,stream>>>(bufH, Wh, b1a, bufY, (int)rows1);
    k_stats<<<B*32,256,0,stream>>>(bufY, meanb, rstdb, S1*K, 32);
    k_act_f16<<<ceilDiv(rows1*32,256),256,0,stream>>>(bufY, meanb, rstdb, bufH, rows1*32, 32, S1*K);

    k_w_to_f16<<<ceilDiv(32*32,256),256,0,stream>>>(w1b, Wh, 32, 32, 32);
    k_gemm_wmma<32,32,4><<<ceilDiv(rows1/64,8),256,0,stream>>>(bufH, Wh, b1b, bufY, (int)rows1);
    k_stats<<<B*32,256,0,stream>>>(bufY, meanb, rstdb, S1*K, 32);
    k_act_f16<<<ceilDiv(rows1*32,256),256,0,stream>>>(bufY, meanb, rstdb, bufH, rows1*32, 32, S1*K);

    k_w_to_f16<<<ceilDiv(32*32,256),256,0,stream>>>(w1c, Wh, 32, 32, 32);
    k_gemm_wmma<32,32,4><<<ceilDiv(rows1/64,8),256,0,stream>>>(bufH, Wh, b1c, bufY, (int)rows1);
    k_stats<<<B*32,256,0,stream>>>(bufY, meanb, rstdb, S1*K, 32);
    k_nrm_pool<<<ceilDiv((long long)B*S1*32,256),256,0,stream>>>(bufY, meanb, rstdb,
                                                     f1, nullptr, B, S1, K, 32);

    // ---------------- SA2 ----------------
    k_fps<<<B,1024,(size_t)S1*16,stream>>>(nxyz1, out_idx2, S1, S2);   // 64 KB LDS
    k_gather_xyz<<<ceilDiv((long long)B*S2,256),256,0,stream>>>(nxyz1, out_idx2, nxyz2, out_pc2, B, S1, S2);
    k_knn<<<B*S2,32,0,stream>>>(nxyz2, nxyz1, nidx2, S2, S1, K);
    k_group_f16<<<ceilDiv(rows2,256),256,0,stream>>>(nxyz1, f1, nxyz2, nidx2, bufH, B, S2, S1, K, 32, 64);

    k_w_to_f16<<<ceilDiv(64*64,256),256,0,stream>>>(w2a, Wh, 64, 35, 64);
    k_gemm_wmma<64,64,2><<<ceilDiv(rows2/32,8),256,0,stream>>>(bufH, Wh, b2a, bufY, (int)rows2);
    k_stats<<<B*64,256,0,stream>>>(bufY, meanb, rstdb, S2*K, 64);
    k_act_f16<<<ceilDiv(rows2*64,256),256,0,stream>>>(bufY, meanb, rstdb, bufH, rows2*64, 64, S2*K);

    k_w_to_f16<<<ceilDiv(64*64,256),256,0,stream>>>(w2b, Wh, 64, 64, 64);
    k_gemm_wmma<64,64,2><<<ceilDiv(rows2/32,8),256,0,stream>>>(bufH, Wh, b2b, bufY, (int)rows2);
    k_stats<<<B*64,256,0,stream>>>(bufY, meanb, rstdb, S2*K, 64);
    k_act_f16<<<ceilDiv(rows2*64,256),256,0,stream>>>(bufY, meanb, rstdb, bufH, rows2*64, 64, S2*K);

    k_w_to_f16<<<ceilDiv(64*64,256),256,0,stream>>>(w2c, Wh, 64, 64, 64);
    k_gemm_wmma<64,64,2><<<ceilDiv(rows2/32,8),256,0,stream>>>(bufH, Wh, b2c, bufY, (int)rows2);
    k_stats<<<B*64,256,0,stream>>>(bufY, meanb, rstdb, S2*K, 64);
    k_nrm_pool<<<ceilDiv((long long)B*S2*64,256),256,0,stream>>>(bufY, meanb, rstdb,
                                                     nullptr, out_f2, B, S2, K, 64);
}
